// GraphEncoder_43018392437384
// MI455X (gfx1250) — matrix operand order repaired and checked by reading, not compiled, not run on hardware
//
#include <hip/hip_runtime.h>
#include <hip/hip_bf16.h>

#define D 128
#define LAYERS 3
#define BN_EPS 1e-5f

typedef __attribute__((ext_vector_type(16))) __bf16        v16bf;
typedef __attribute__((ext_vector_type(8)))  float         v8f;
typedef __attribute__((ext_vector_type(8)))  unsigned int  v8u;
typedef __attribute__((ext_vector_type(4)))  unsigned int  u32x4;
typedef __attribute__((ext_vector_type(8)))  unsigned int  u32x8;

// ---------------- degree / norm ----------------
__global__ __launch_bounds__(256) void deg_init_k(float* deg, int n) {
    int i = blockIdx.x * 256 + threadIdx.x;
    if (i < n) deg[i] = 1.0f;               // self-loop weight
}

__global__ __launch_bounds__(256) void deg_edges_k(const int* __restrict__ dst,
                                                   const float* __restrict__ ew,
                                                   float* deg, int e) {
    int i = blockIdx.x * 256 + threadIdx.x;
    if (i < e) unsafeAtomicAdd(&deg[dst[i]], ew[i]);
}

__global__ __launch_bounds__(256) void dinv_k(const float* __restrict__ deg,
                                              float* dinv, int n) {
    int i = blockIdx.x * 256 + threadIdx.x;
    if (i < n) {
        float d = deg[i];
        dinv[i] = (d > 0.0f) ? rsqrtf(fmaxf(d, 1e-12f)) : 0.0f;
    }
}

// ---------------- fp32 -> bf16 cast of initial features ----------------
__global__ __launch_bounds__(256) void xcast_k(const float* __restrict__ x,
                                               unsigned short* __restrict__ hbf,
                                               size_t total) {
    size_t i = (size_t)blockIdx.x * 256 + threadIdx.x;
    if (i < total)
        hbf[i] = (unsigned short)(__builtin_bit_cast(unsigned int, x[i]) >> 16);
}

// ---------------- WMMA GEMM: out[N,128] = hbf[N,128](bf16) @ W[128,128] ----------
// Block: 256 threads (8 waves). Block computes a 128-row x 128-col panel.
// A panel (32 KB contiguous bf16) staged via Tensor Data Mover; W staged
// transposed+converted by the block. LDS total: 64 KB.
__global__ __launch_bounds__(256) void gemm_bf16_k(const unsigned int* __restrict__ hbf,
                                                   const float* __restrict__ W,
                                                   float* __restrict__ out,
                                                   int nrows) {
    __shared__ unsigned short sWt[D * D];      // Wt[n][k]  (transposed), 32 KB
    __shared__ unsigned int   sA[D * (D / 2)]; // A[m][k/2] bf16 pairs,   32 KB

    const int tid       = threadIdx.x;
    const int block_row = blockIdx.x * 128;

    // Stage W transposed: W is row-major [k][n]; store bf16 at Wt[n*128 + k]
    for (int i = tid; i < D * D; i += 256) {
        int k = i >> 7, n = i & 127;
        unsigned int bits = __builtin_bit_cast(unsigned int, W[i]);
        sWt[n * D + k] = (unsigned short)(bits >> 16);
    }

    // Stage A panel via TDM: one wave issues a 1-D tensor_load_to_lds of
    // 8192 dwords (32 KB) from hbf + block_row*256B into sA.
    if (tid < 32) {
        unsigned long long ga = (unsigned long long)(uintptr_t)hbf
                              + (unsigned long long)block_row * (D * 2);
        unsigned int lds_off = (unsigned int)(uintptr_t)(void*)&sA[0]; // LDS byte offset
        unsigned int rem_dw  = (unsigned int)nrows * 64u
                             - (unsigned int)block_row * 64u;          // dwords left

        u32x4 g0;
        g0[0] = 1u;                                   // count=1, user descriptor
        g0[1] = lds_off;                              // lds_addr
        g0[2] = (unsigned int)(ga & 0xFFFFFFFFull);   // global_addr[31:0]
        g0[3] = (unsigned int)((ga >> 32) & 0x1FFFFFFull) | (2u << 30); // [56:32]|type=2

        u32x8 g1;
        g1[0] = (2u << 16);                           // wgmask=0, data_size=2 (4B)
        g1[1] = (rem_dw & 0xFFFFu) << 16;             // tensor_dim0[15:0]
        g1[2] = (rem_dw >> 16) | (1u << 16);          // tensor_dim0[31:16] | tensor_dim1=1
        g1[3] = (8192u << 16);                        // tensor_dim1 hi=0 | tile_dim0=8192
        g1[4] = 0u;                                   // tile_dim1=0, tile_dim2=0 (1-D tile)
        g1[5] = rem_dw;                               // tensor_dim0_stride[31:0]
        g1[6] = 0u;                                   // stride hi bits
        g1[7] = 0u;

        asm volatile("tensor_load_to_lds %0, %1" :: "s"(g0), "s"(g1) : "memory");
        __builtin_amdgcn_s_wait_tensorcnt(0);
    }
    __syncthreads();

    const int wave = tid >> 5;     // 0..7 : 16-row strip
    const int lane = tid & 31;
    const int l15  = lane & 15;
    const int l16  = lane >> 4;

    v8f acc[8];
    v8f zero = {0.f, 0.f, 0.f, 0.f, 0.f, 0.f, 0.f, 0.f};
#pragma unroll
    for (int j = 0; j < 8; j++) acc[j] = zero;

    const unsigned int* sWt32 = (const unsigned int*)sWt; // row = 64 dwords

#pragma unroll
    for (int kk = 0; kk < D; kk += 32) {
        // A fragment (16x32 bf16), ISA 7.12.2 layout
        v8u af;
#pragma unroll
        for (int v = 0; v < 8; v++) {
            int k0 = kk + ((v < 4) ? (l16 * 8 + v * 2)
                                   : (16 + l16 * 8 + (v - 4) * 2));
            af[v] = sA[(wave * 16 + l15) * 64 + (k0 >> 1)];
        }
        v16bf a = __builtin_bit_cast(v16bf, af);

#pragma unroll
        for (int j = 0; j < 8; j++) {
            // B fragment (32x16 bf16): same k order as A (ISA 7.12.2 table layout)
            v8u bfu;
#pragma unroll
            for (int v = 0; v < 8; v++) {
                int k = kk + ((v < 4) ? (l16 * 8 + v * 2)
                                      : (16 + l16 * 8 + (v - 4) * 2));
                bfu[v] = sWt32[(j * 16 + l15) * 64 + (k >> 1)];
            }
            v16bf b = __builtin_bit_cast(v16bf, bfu);
            acc[j] = __builtin_amdgcn_wmma_f32_16x16x32_bf16(
                false, a, false, b, (short)0, acc[j], false, false);
        }
    }

    // Store C/D tile: VGPR v -> M=v (lanes 0-15) / M=v+8 (lanes 16-31), N=lane%16
#pragma unroll
    for (int j = 0; j < 8; j++) {
#pragma unroll
        for (int v = 0; v < 8; v++) {
            int m   = (lane < 16) ? v : (v + 8);
            int row = block_row + wave * 16 + m;
            if (row < nrows) out[(size_t)row * D + j * 16 + l15] = acc[j][v];
        }
    }
}

// ---------------- self-loop term + bias (full overwrite of agg) ----------------
__global__ __launch_bounds__(256) void agg_init_k(const float* __restrict__ hw,
                                                  const float* __restrict__ dinv,
                                                  const float* __restrict__ bias,
                                                  float* __restrict__ agg,
                                                  size_t total) {
    size_t i = (size_t)blockIdx.x * 256 + threadIdx.x;
    if (i < total) {
        int row = (int)(i >> 7);
        int c   = (int)(i & 127);
        float di = dinv[row];
        agg[i] = di * di * hw[i] + bias[c];
    }
}

// ---------------- edge scatter: one wave32 per edge, 4 cols/lane ----------------
__global__ __launch_bounds__(256) void scatter_k(const int* __restrict__ src,
                                                 const int* __restrict__ dst,
                                                 const float* __restrict__ ew,
                                                 const float* __restrict__ dinv,
                                                 const float* __restrict__ hw,
                                                 float* __restrict__ agg,
                                                 int nedges) {
    int wid  = blockIdx.x * 8 + (threadIdx.x >> 5);
    int lane = threadIdx.x & 31;
    if (wid >= nedges) return;

    int s = src[wid], d = dst[wid];
    if (lane == 0 && wid + 64 < nedges)
        __builtin_prefetch(hw + (size_t)src[wid + 64] * D, 0, 0);

    float coef = dinv[s] * ew[wid] * dinv[d];
    float4 v = *(const float4*)(hw + (size_t)s * D + lane * 4);
    float* op = agg + (size_t)d * D + lane * 4;
    unsafeAtomicAdd(op + 0, coef * v.x);
    unsafeAtomicAdd(op + 1, coef * v.y);
    unsafeAtomicAdd(op + 2, coef * v.z);
    unsafeAtomicAdd(op + 3, coef * v.w);
}

// ---------------- zero BN accumulators ----------------
__global__ __launch_bounds__(256) void zero_stats_k(float* stats) {
    stats[threadIdx.x] = 0.0f;
}

// ---------------- fused skip + ReLU + BN column stats ----------------
__global__ __launch_bounds__(256) void skip_stats_k(const float* __restrict__ agg,
                                                    const float* __restrict__ x0,
                                                    float* __restrict__ tmp,
                                                    float* __restrict__ ssum,
                                                    float* __restrict__ ssum2,
                                                    int relu, int nrows) {
    __shared__ float ls[256];
    __shared__ float ls2[256];
    const int tid  = threadIdx.x;
    const int c    = tid & 127;
    const int half = tid >> 7;
    const int base = blockIdx.x * 64;

    float s = 0.0f, s2 = 0.0f;
    for (int i = half; i < 64; i += 2) {
        int r = base + i;
        if (r < nrows) {
            size_t idx = (size_t)r * D + c;
            float v = agg[idx] + x0[idx];
            if (relu) v = fmaxf(v, 0.0f);
            tmp[idx] = v;
            s += v;
            s2 += v * v;
        }
    }
    ls[tid] = s; ls2[tid] = s2;
    __syncthreads();
    if (tid < 128) {
        unsafeAtomicAdd(&ssum[tid],  ls[tid]  + ls[tid + 128]);
        unsafeAtomicAdd(&ssum2[tid], ls2[tid] + ls2[tid + 128]);
    }
}

// ---------------- BN apply: bf16 activations for next layer (+fp32 final out) --
__global__ __launch_bounds__(256) void bn_apply_k(const float* __restrict__ tmp,
                                                  const float* __restrict__ ssum,
                                                  const float* __restrict__ ssum2,
                                                  const float* __restrict__ gamma,
                                                  const float* __restrict__ beta,
                                                  float* __restrict__ outf,
                                                  int write_f32,
                                                  unsigned short* __restrict__ hbf,
                                                  size_t total, float invN) {
    size_t i = (size_t)blockIdx.x * 256 + threadIdx.x;
    if (i < total) {
        int c = (int)(i & 127);
        float mean = ssum[c] * invN;
        float var  = ssum2[c] * invN - mean * mean;
        float v = (tmp[i] - mean) * rsqrtf(var + BN_EPS) * gamma[c] + beta[c];
        if (write_f32) outf[i] = v;
        hbf[i] = (unsigned short)(__builtin_bit_cast(unsigned int, v) >> 16);
    }
}

// ---------------- host launcher ----------------
extern "C" void kernel_launch(void* const* d_in, const int* in_sizes, int n_in,
                              void* d_out, int out_size, void* d_ws, size_t ws_size,
                              hipStream_t stream) {
    const float* x    = (const float*)d_in[0];
    const int*   ei   = (const int*)d_in[1];
    const float* ew   = (const float*)d_in[2];
    const float* Ws   = (const float*)d_in[3];
    const float* bs   = (const float*)d_in[4];
    const float* gams = (const float*)d_in[5];
    const float* bets = (const float*)d_in[6];

    const int N = in_sizes[0] / D;
    const int E = in_sizes[1] / 2;
    const int* srcI = ei;
    const int* dstI = ei + E;

    // workspace layout
    float* hw    = (float*)d_ws;                 // N*D fp32 (GEMM out / BN tmp)
    float* agg   = hw   + (size_t)N * D;         // N*D fp32 (aggregation)
    float* deg   = agg  + (size_t)N * D;         // N
    float* dinv  = deg  + N;                     // N
    float* stats = dinv + N;                     // 256 (ssum | ssum2)
    unsigned short* hbf = (unsigned short*)(stats + 256); // N*D bf16 activations

    const float invN = 1.0f / (float)N;
    const size_t total = (size_t)N * D;
    const int nbN    = (N + 255) / 256;
    const int nbE    = (E + 255) / 256;
    const int nbElem = (int)((total + 255) / 256);
    const int nbGemm = (N + 127) / 128;
    const int nbScat = (E + 7) / 8;
    const int nbStat = (N + 63) / 64;

    deg_init_k <<<nbN, 256, 0, stream>>>(deg, N);
    deg_edges_k<<<nbE, 256, 0, stream>>>(dstI, ew, deg, E);
    dinv_k     <<<nbN, 256, 0, stream>>>(deg, dinv, N);
    xcast_k    <<<nbElem, 256, 0, stream>>>(x, hbf, total);

    for (int l = 0; l < LAYERS; l++) {
        gemm_bf16_k<<<nbGemm, 256, 0, stream>>>((const unsigned int*)hbf,
                                                Ws + (size_t)l * D * D, hw, N);
        agg_init_k <<<nbElem, 256, 0, stream>>>(hw, dinv, bs + l * D, agg, total);
        scatter_k  <<<nbScat, 256, 0, stream>>>(srcI, dstI, ew, dinv, hw, agg, E);
        zero_stats_k<<<1, 256, 0, stream>>>(stats);
        skip_stats_k<<<nbStat, 256, 0, stream>>>(agg, x, hw, stats, stats + D,
                                                 (l < LAYERS - 1) ? 1 : 0, N);
        bn_apply_k<<<nbElem, 256, 0, stream>>>(hw, stats, stats + D,
                                               gams + l * D, bets + l * D,
                                               (float*)d_out,
                                               (l == LAYERS - 1) ? 1 : 0,
                                               hbf, total, invN);
    }
}
